// KroneNet_59485297049802
// MI455X (gfx1250) — matrix operand order repaired and checked by reading, not compiled, not run on hardware
//
#include <hip/hip_runtime.h>
#include <hip/hip_bf16.h>

typedef __attribute__((ext_vector_type(16))) _Float16 v16h;
typedef __attribute__((ext_vector_type(8)))  float    v8f;

#define BLOCK 256
#define WAVES_PER_BLOCK 8

// A-matrix (16x32 f16) per-lane K pattern, per CDNA5 ISA 7.12.2:
// lanes 0-15: V0..V3 -> K=0..7, V4..V7 -> K=16..23
// lanes 16-31: V0..V3 -> K=8..15, V4..V7 -> K=24..31
__device__ __forceinline__ int kpatA(int lane, int s) {
    int v = s >> 1;
    int base = (lane & 16) ? 8 : 0;
    int kk = (v < 4) ? (v << 1) : (16 + ((v - 4) << 1));
    return base + kk + (s & 1);
}

// One MLP chain: layer1 (K=2) built directly into WMMA A-fragments,
// layer2 via 32 WMMAs (8 N-tiles x 4 K-chunks), layer3 partial sums in D layout.
// NOUT = number of layer-3 output columns (1 for chain A, 3 for chain B).
template <int NOUT>
__device__ __forceinline__ void run_chain(
    int lane, float2 xv,
    const float2* __restrict__ w1_2, const float* __restrict__ b1,
    const _Float16* w2frag,          // LDS fragment base for this chain
    const float* __restrict__ b2, const float* __restrict__ w3,
    float (&p)[NOUT][8])
{
    // Build A fragments (16x32 f16 each) for K = 0..127.
    v16h a[4];
    #pragma unroll
    for (int kc = 0; kc < 4; ++kc) {
        #pragma unroll
        for (int s = 0; s < 16; ++s) {
            int k = kc * 32 + kpatA(lane, s);
            float2 w = w1_2[k];
            float v = fmaf(xv.y, w.y, fmaf(xv.x, w.x, b1[k]));
            a[kc][s] = (_Float16)fmaxf(v, 0.0f);
        }
    }

    #pragma unroll
    for (int t = 0; t < NOUT; ++t)
        #pragma unroll
        for (int j = 0; j < 8; ++j) p[t][j] = 0.0f;

    const int m = lane & 15;
    // Keep this loop rolled: small scheduling region -> LDS loads stay next to
    // their WMMA and nothing spills to scratch.
    #pragma unroll 1
    for (int nt = 0; nt < 8; ++nt) {
        const int n = nt * 16 + m;
        const float bias = b2[n];
        v8f acc;
        #pragma unroll
        for (int j = 0; j < 8; ++j) acc[j] = bias;

        #pragma unroll
        for (int kc = 0; kc < 4; ++kc) {
            const v16h bf = *(const v16h*)(w2frag + (nt * 4 + kc) * 512 + lane * 16);
            acc = __builtin_amdgcn_wmma_f32_16x16x32_f16(
                false, a[kc], false, bf, (short)0, acc, false, false);
        }

        float w3v[NOUT];
        #pragma unroll
        for (int t = 0; t < NOUT; ++t) w3v[t] = w3[t * 128 + n];

        #pragma unroll
        for (int j = 0; j < 8; ++j) {
            float r = fmaxf(acc[j], 0.0f);   // relu after layer 2
            #pragma unroll
            for (int t = 0; t < NOUT; ++t) p[t][j] = fmaf(r, w3v[t], p[t][j]);
        }
    }
}

__global__ __launch_bounds__(BLOCK) void krone_kernel(
    const float* __restrict__ x,
    const float* __restrict__ w1a, const float* __restrict__ w1b,
    const float* __restrict__ b1a, const float* __restrict__ b1b,
    const float* __restrict__ w2a, const float* __restrict__ w2b,
    const float* __restrict__ b2a, const float* __restrict__ b2b,
    const float* __restrict__ w3a, const float* __restrict__ w3b,
    const float* __restrict__ b3a, const float* __restrict__ b3b,
    float* __restrict__ out, int Brows)
{
    // w2 staged as f16 in B-fragment order: flat = c*16384 + nt*2048 + kc*512 + lane*16 + s
    // B 32x16 layout: N = lane%16 ; K = kc*32 + (lane<16 ? 0 : 8) + s + (s<8 ? 0 : 8)
    __shared__ _Float16 w2lds[32768];   // exactly 64 KB

    for (int flat = threadIdx.x; flat < 32768; flat += BLOCK) {
        int s    = flat & 15;
        int lane = (flat >> 4)  & 31;
        int kc   = (flat >> 9)  & 3;
        int nt   = (flat >> 11) & 7;
        int c    = (flat >> 14) & 1;
        int k = kc * 32 + ((lane & 16) ? 8 : 0) + s + ((s & 8) ? 8 : 0);
        int n = nt * 16 + (lane & 15);
        const float* w2 = c ? w2b : w2a;
        w2lds[flat] = (_Float16)w2[n * 128 + k];   // B[k][n] = w2[n][k]
    }
    __syncthreads();

    const int lane   = threadIdx.x & 31;
    const int wid    = blockIdx.x * WAVES_PER_BLOCK + (threadIdx.x >> 5);
    const int nwaves = gridDim.x * WAVES_PER_BLOCK;
    const int ntiles = Brows >> 4;

    const float2* x2a  = (const float2*)x;
    const float2* x2b  = (const float2*)(x + (size_t)Brows * 2);
    const float2* w1a2 = (const float2*)w1a;
    const float2* w1b2 = (const float2*)w1b;

    #pragma unroll 1
    for (int tile = wid; tile < ntiles; tile += nwaves) {
        const int m   = lane & 15;
        const int row = (tile << 4) + m;

        // Chain A (layer-3 has 1 output), then chain B (3 outputs) — sequential
        // so only one chain's A-fragments + accumulator are live at a time.
        float pa[1][8];
        run_chain<1>(lane, x2a[row], w1a2, b1a, w2lds,         b2a, w3a, pa);
        float pb[3][8];
        run_chain<3>(lane, x2b[row], w1b2, b1b, w2lds + 16384, b2b, w3b, pb);

        // Reduce across the 16 lanes of each half (N dimension of D layout).
        #pragma unroll
        for (int mask = 1; mask <= 8; mask <<= 1) {
            #pragma unroll
            for (int j = 0; j < 8; ++j) {
                pa[0][j] += __shfl_xor(pa[0][j], mask, 32);
                pb[0][j] += __shfl_xor(pb[0][j], mask, 32);
                pb[1][j] += __shfl_xor(pb[1][j], mask, 32);
                pb[2][j] += __shfl_xor(pb[2][j], mask, 32);
            }
        }

        // Lane 0 holds M=0..7, lane 16 holds M=8..15: finish layer 3 + softmax(3).
        if ((lane & 15) == 0) {
            const int mbase = (tile << 4) + ((lane >> 4) << 3);
            const float ba  = b3a[0];
            const float bb0 = b3b[0], bb1 = b3b[1], bb2 = b3b[2];
            #pragma unroll
            for (int j = 0; j < 8; ++j) {
                float sa = pa[0][j] + ba;              // no relu after layer 3
                float y0 = sa * (pb[0][j] + bb0);
                float y1 = sa * (pb[1][j] + bb1);
                float y2 = sa * (pb[2][j] + bb2);
                float mx = fmaxf(y0, fmaxf(y1, y2));
                float e0 = __expf(y0 - mx);
                float e1 = __expf(y1 - mx);
                float e2 = __expf(y2 - mx);
                float inv = 1.0f / (e0 + e1 + e2);
                float* o = out + (size_t)(mbase + j) * 3;
                o[0] = e0 * inv;
                o[1] = e1 * inv;
                o[2] = e2 * inv;
            }
        }
    }
}

extern "C" void kernel_launch(void* const* d_in, const int* in_sizes, int n_in,
                              void* d_out, int out_size, void* d_ws, size_t ws_size,
                              hipStream_t stream) {
    const float* x   = (const float*)d_in[0];
    const float* w1a = (const float*)d_in[1];
    const float* w1b = (const float*)d_in[2];
    const float* b1a = (const float*)d_in[3];
    const float* b1b = (const float*)d_in[4];
    const float* w2a = (const float*)d_in[5];
    const float* w2b = (const float*)d_in[6];
    const float* b2a = (const float*)d_in[7];
    const float* b2b = (const float*)d_in[8];
    const float* w3a = (const float*)d_in[9];
    const float* w3b = (const float*)d_in[10];
    const float* b3a = (const float*)d_in[11];
    const float* b3b = (const float*)d_in[12];

    const int Brows  = in_sizes[0] / 4;     // x is (2, B, 2) f32
    const int ntiles = Brows >> 4;

    int blocks = (ntiles + 31) / 32;        // ~4 tiles per wave at full size
    if (blocks > 512) blocks = 512;
    if (blocks < 1)   blocks = 1;

    krone_kernel<<<blocks, BLOCK, 0, stream>>>(
        x, w1a, w1b, b1a, b1b, w2a, w2b, b2a, b2b,
        w3a, w3b, b3a, b3b, (float*)d_out, Brows);
}
